// TemporalGNN_28724741275827
// MI455X (gfx1250) — compile-verified, matrix operand order checked by reading, not run on hardware
//
#include <hip/hip_runtime.h>
#include <hip/hip_bf16.h>
#include <stdint.h>
#include <stddef.h>

typedef __attribute__((ext_vector_type(16))) _Float16 v16h;
typedef __attribute__((ext_vector_type(8)))  _Float16 v8h;
typedef __attribute__((ext_vector_type(4)))  _Float16 v4h;
typedef __attribute__((ext_vector_type(8)))  float    v8f;
typedef __attribute__((ext_vector_type(4)))  float    v4f;
typedef __attribute__((ext_vector_type(4)))  unsigned int v4u;
typedef __attribute__((ext_vector_type(8)))  int      v8i;
typedef __attribute__((ext_vector_type(4)))  int      v4i;

static constexpr int G = 128;     // B*S graphs
static constexpr int N = 512;     // nodes per graph
static constexpr int F = 64;
static constexpr int H = 128;
static constexpr int O = 64;
static constexpr float BN_EPS = 1e-5f;

static constexpr int ADJ_LDW = 520;   // f16 row stride for adj strip in LDS (260 dwords)
static constexpr int YT_LDW  = 520;   // f16 row stride for Yt panel in LDS (260 dwords, via TDM pad)

// ---------------- workspace layout (bytes) ----------------
static constexpr size_t OFF_DIS   = 0;                           // G*N f32
static constexpr size_t OFF_H0    = OFF_DIS  + (size_t)G*N*4;    // G*H f32
static constexpr size_t OFF_A0    = OFF_H0   + (size_t)G*H*4;    // G*H f32 (z0*scale0)
static constexpr size_t OFF_B0    = OFF_A0   + (size_t)G*H*4;    // H f32
static constexpr size_t OFF_C1S   = OFF_B0   + (size_t)H*4;      // H f32
static constexpr size_t OFF_C1B   = OFF_C1S  + (size_t)H*4;      // H f32
static constexpr size_t OFF_W1T   = OFF_C1B  + (size_t)H*4;      // H*H f16
static constexpr size_t OFF_WOT   = OFF_W1T  + (size_t)H*H*2;    // O*H f16
static constexpr size_t OFF_H1F32 = OFF_WOT  + (size_t)O*H*2;    // G*N*H f32
static constexpr size_t OFF_H1F16 = OFF_H1F32 + (size_t)G*N*H*4; // G*N*H f16
static constexpr size_t OFF_YT    = OFF_H1F16 + (size_t)G*N*H*2; // G*H*N f16 (Y transposed)

__device__ __forceinline__ float wave_sum32(float v) {
    #pragma unroll
    for (int off = 16; off > 0; off >>= 1) v += __shfl_xor(v, off, 32);
    return v;
}

__device__ __forceinline__ uint32_t lds_off(const void* p) {
    return (uint32_t)(size_t)(__attribute__((address_space(3))) const void*)p;
}

// f16 fragment loader (global): element (r,k) at base[r*ld + k].
__device__ __forceinline__ v16h frag_ld_f16(const _Float16* base, int ld, int lane) {
    const int r  = lane & 15;
    const int h8 = (lane >> 4) * 8;
    const _Float16* p = base + (size_t)r * ld + h8;
    v8h lo = *(const v8h*)p;
    v8h hi = *(const v8h*)(p + 16);
    v16h f;
    #pragma unroll
    for (int i = 0; i < 8; ++i) { f[i] = lo[i]; f[i + 8] = hi[i]; }
    return f;
}

// ---------------- kernel 1a: h0 = x @ W_in + b_in ----------------
__global__ void k_h0(const float* __restrict__ x, const float* __restrict__ Win,
                     const float* __restrict__ bin, float* __restrict__ h0) {
    __shared__ float sx[F];
    const int g = blockIdx.x, h = threadIdx.x;
    if (h < F) sx[h] = x[(size_t)g * F + h];
    __syncthreads();
    float acc = bin[h];
    #pragma unroll 8
    for (int f = 0; f < F; ++f) acc += sx[f] * Win[(size_t)f * H + h];
    h0[(size_t)g * H + h] = acc;
}

// ---------------- kernel 1b: A0 = (h0 @ Wc0) * bn_scale0 ----------------
__global__ void k_a0(const float* __restrict__ h0, const float* __restrict__ Wconv,
                     const float* __restrict__ gamma, const float* __restrict__ rv,
                     float* __restrict__ A0) {
    __shared__ float sh[H];
    const int g = blockIdx.x, h = threadIdx.x;
    sh[h] = h0[(size_t)g * H + h];
    __syncthreads();
    float z = 0.f;
    #pragma unroll 8
    for (int k = 0; k < H; ++k) z += sh[k] * Wconv[(size_t)k * H + h];
    const float sc0 = gamma[h] * rsqrtf(rv[h] + BN_EPS);
    A0[(size_t)g * H + h] = z * sc0;
}

// ---------------- kernel 1c: fused BN consts + transposed f16 weights ----------------
__global__ void k_consts(const float* __restrict__ Wconv, const float* __restrict__ bconv,
                         const float* __restrict__ gamma, const float* __restrict__ beta,
                         const float* __restrict__ rm, const float* __restrict__ rv,
                         const float* __restrict__ Wout,
                         float* __restrict__ B0, float* __restrict__ C1s, float* __restrict__ C1b,
                         _Float16* __restrict__ W1T, _Float16* __restrict__ WoT) {
    const int idx = blockIdx.x * blockDim.x + threadIdx.x;
    if (idx < H * H) {
        const int k = idx / H, h = idx % H;
        W1T[(size_t)h * H + k] = (_Float16)Wconv[(size_t)(H + k) * H + h];
    }
    if (idx < O * H) {
        const int o = idx / H, k = idx % H;
        WoT[(size_t)o * H + k] = (_Float16)Wout[(size_t)k * O + o];
    }
    if (idx < H) {
        const float s0 = gamma[idx] * rsqrtf(rv[idx] + BN_EPS);
        B0[idx]  = bconv[idx] * s0 + beta[idx] - rm[idx] * s0;
        const float s1 = gamma[H + idx] * rsqrtf(rv[H + idx] + BN_EPS);
        C1s[idx] = s1;
        C1b[idx] = bconv[H + idx] * s1 + beta[H + idx] - rm[H + idx] * s1;
    }
}

// ---------------- kernel 2: dis = rsqrt(1 + rowsum(adj)) ----------------
__global__ void k_deg(const float* __restrict__ adj, float* __restrict__ dis) {
    const int lane = threadIdx.x & 31, wave = threadIdx.x >> 5;
    const int row = blockIdx.x * 8 + wave;
    const float* p = adj + (size_t)row * N;
    float s = 0.f;
    #pragma unroll
    for (int it = 0; it < 4; ++it) {
        v4f a = *(const v4f*)(p + it * 128 + lane * 4);
        s += a[0] + a[1] + a[2] + a[3];
    }
    s = wave_sum32(s);
    if (lane == 0) dis[row] = rsqrtf(1.0f + s);
}

// ---------------- kernel 3: layer-1 rowsum shortcut -> h1 (f32 + f16) ----------------
__global__ void k_layer1(const float* __restrict__ adj, const float* __restrict__ dis,
                         const float* __restrict__ h0, const float* __restrict__ A0,
                         const float* __restrict__ B0,
                         float* __restrict__ h1f, _Float16* __restrict__ h1h) {
    const int lane = threadIdx.x & 31, wave = threadIdx.x >> 5;
    const int row = blockIdx.x * 8 + wave;
    const int g = row >> 9;
    const float* ap = adj + (size_t)row * N;
    const float* dp = dis + (size_t)g * N;
    float s = 0.f;
    #pragma unroll
    for (int it = 0; it < 4; ++it) {
        v4f a = *(const v4f*)(ap + it * 128 + lane * 4);
        v4f d = *(const v4f*)(dp + it * 128 + lane * 4);
        s += a[0]*d[0] + a[1]*d[1] + a[2]*d[2] + a[3]*d[3];
    }
    s = wave_sum32(s);
    const float di = dis[row];
    const float rs = di * (di + s);
    #pragma unroll
    for (int k = 0; k < 4; ++k) {
        const int h = lane + k * 32;
        float v = rs * A0[(size_t)g * H + h] + B0[h];
        v = fmaxf(v, 0.f) + h0[(size_t)g * H + h];
        h1f[(size_t)row * H + h] = v;
        h1h[(size_t)row * H + h] = (_Float16)v;
    }
}

// ---------------- kernel 3b: Yt[g][h][j] = (h1 @ Wc1)[j][h] via WMMA ----------------
__global__ void k_ytrans(const _Float16* __restrict__ W1T, const _Float16* __restrict__ h1h,
                         _Float16* __restrict__ yt) {
    const int g = blockIdx.y, j0 = blockIdx.x * 16;
    const int lane = threadIdx.x & 31, wave = threadIdx.x >> 5;
    const int m0 = wave * 16;
    v8f c = {};
    #pragma unroll
    for (int kk = 0; kk < H; kk += 32) {
        v16h a = frag_ld_f16(W1T + (size_t)m0 * H + kk, H, lane);
        v16h b = frag_ld_f16(h1h + ((size_t)g * N + j0) * H + kk, H, lane);
        c = __builtin_amdgcn_wmma_f32_16x16x32_f16(false, a, false, b, (short)0, c, false, false);
    }
    const int col = lane & 15, hh = (lane >> 4) * 8;
    #pragma unroll
    for (int r = 0; r < 8; ++r) {
        const int mr = m0 + r + hh;
        yt[((size_t)g * H + mr) * N + j0 + col] = (_Float16)c[r];
    }
}

// ---------------- kernel 4: layer-2 SpMM + BN/ReLU/residual + out-proj ----------------
// Grid (4, G): block owns 8 row-strips of one graph. Yt panel staged once per
// block via Tensor Data Mover (with LDS padding); adjacency strips normalized
// cooperatively into LDS f16; dis staged via async-to-LDS.
__global__ void k_layer2(const float* __restrict__ adj, const float* __restrict__ dis,
                         const _Float16* __restrict__ yt, const float* __restrict__ h1f,
                         const float* __restrict__ C1s, const float* __restrict__ C1b,
                         const _Float16* __restrict__ WoT, const float* __restrict__ bout,
                         const float* __restrict__ mask, float* __restrict__ node) {
    __shared__ float    s_dis[N];                 // 2 KB
    __shared__ _Float16 s_yt[H * YT_LDW];         // 130 KB (TDM pads rows to 260 dwords)
    __shared__ _Float16 s_adjh[16 * ADJ_LDW];     // 16.25 KB normalized adjacency strip
    __shared__ _Float16 s_h2[16 * 136];           // h2 strip for out-projection
    const int g = blockIdx.y, i_base = blockIdx.x * 128;
    const int tid = threadIdx.x, lane = tid & 31, wave = tid >> 5;
    const int r  = lane & 15;
    const int h8 = (lane >> 4) * 8;

    // ---- stage dis[] via gfx1250 async-to-LDS (128 x b128) ----
    if (tid < 128) {
        uint32_t lo = lds_off(&s_dis[0]) + (uint32_t)tid * 16u;
        uint64_t ga = (uint64_t)(const void*)(dis + (size_t)g * N + tid * 4);
        asm volatile("global_load_async_to_lds_b128 %0, %1, off" :: "v"(lo), "v"(ga) : "memory");
    }
    asm volatile("s_wait_asynccnt 0x0" ::: "memory");

    // ---- stage the whole Yt panel [H x N] f16 via the Tensor Data Mover ----
    if (wave == 0) {
        const uint64_t ga = (uint64_t)(const void*)(yt + (size_t)g * H * N);
        v4u g0;
        g0[0] = 1u;                                            // count=1, user mode
        g0[1] = lds_off(&s_yt[0]);                             // lds_addr
        g0[2] = (uint32_t)ga;                                  // global_addr[31:0]
        g0[3] = (uint32_t)((ga >> 32) & 0x01FFFFFFu) | (2u << 30);  // addr[56:32] | type=2
        v8i g1;
        g1[0] = (int)((2u << 16) | (1u << 20) | (7u << 22) | (3u << 25));
        //        data_size=4B   pad_enable   every 256 dw    pad 4 dwords
        g1[1] = (int)(256u << 16);    // tensor_dim0 = 256 dwords per row
        g1[2] = (int)(128u << 16);    // tensor_dim1 = 128 rows
        g1[3] = (int)(256u << 16);    // tile_dim0 = 256 dwords
        g1[4] = (int)128u;            // tile_dim1 = 128 rows, tile_dim2 = 0
        g1[5] = (int)256u;            // tensor_dim0_stride = 256 dwords
        g1[6] = 0; g1[7] = 0;
        v4i z4 = {0, 0, 0, 0};
        v8i z8 = {0, 0, 0, 0, 0, 0, 0, 0};
        __builtin_amdgcn_tensor_load_to_lds(g0, g1, z4, z4, z8, 0);
        __builtin_amdgcn_s_wait_tensorcnt(0);
    }
    __syncthreads();

    for (int strip = 0; strip < 8; ++strip) {
        const int i0 = i_base + strip * 16;
        __syncthreads();   // prev aggregation (s_adjh) and out-proj (s_h2) done

        // ---- cooperative normalize: adj strip -> f16 in LDS ----
        {
            const int row = tid >> 4;
            const int cb  = (tid & 15) * 4;
            const float* src = adj + ((size_t)g * N + i0 + row) * N;
            const float di = s_dis[i0 + row];
            #pragma unroll
            for (int it = 0; it < 8; ++it) {
                const int c = cb + it * 64;
                v4f a = *(const v4f*)(src + c);
                v4h hv;
                #pragma unroll
                for (int k = 0; k < 4; ++k) {
                    float t = a[k] + ((i0 + row == c + k) ? 1.f : 0.f);
                    hv[k] = (_Float16)(t * di * s_dis[c + k]);
                }
                *(v4h*)&s_adjh[row * ADJ_LDW + c] = hv;
            }
        }
        __syncthreads();

        // ---- aggregation: C[16x16] over h-tile = wave*16, K loop over j (all LDS) ----
        const int hc0 = wave * 16;
        v8f acc = {};
        #pragma unroll 4
        for (int jb = 0; jb < N; jb += 32) {
            const _Float16* pa = &s_adjh[r * ADJ_LDW + jb + h8];
            v8h alo = *(const v8h*)pa;
            v8h ahi = *(const v8h*)(pa + 16);
            const _Float16* pb = &s_yt[(hc0 + r) * YT_LDW + jb + h8];
            v8h blo = *(const v8h*)pb;
            v8h bhi = *(const v8h*)(pb + 16);
            v16h af, bf;
            #pragma unroll
            for (int i = 0; i < 8; ++i) {
                af[i] = alo[i]; af[i + 8] = ahi[i];
                bf[i] = blo[i]; bf[i + 8] = bhi[i];
            }
            acc = __builtin_amdgcn_wmma_f32_16x16x32_f16(false, af, false, bf, (short)0, acc, false, false);
        }

        // ---- epilogue: BN + ReLU + residual -> h2 strip (f16) in LDS ----
        {
            const int col = lane & 15, hcol = hc0 + col;
            const float sc = C1s[hcol], sb = C1b[hcol];
            #pragma unroll
            for (int rr = 0; rr < 8; ++rr) {
                const int row = rr + h8;
                float v = fmaxf(acc[rr] * sc + sb, 0.f);
                v += h1f[((size_t)g * N + i0 + row) * H + hcol];
                s_h2[row * 136 + hcol] = (_Float16)v;
            }
        }
        __syncthreads();

        // ---- out-projection: node = h2 @ W_out + b_out (waves 0..3) ----
        if (wave < 4) {
            const int o0 = wave * 16;
            v8f oc = {};
            #pragma unroll
            for (int kk = 0; kk < H; kk += 32) {
                const _Float16* pa = &s_h2[r * 136 + kk + h8];
                v8h lo = *(const v8h*)pa;
                v8h hi = *(const v8h*)(pa + 16);
                v16h af;
                #pragma unroll
                for (int i = 0; i < 8; ++i) { af[i] = lo[i]; af[i + 8] = hi[i]; }
                v16h bf = frag_ld_f16(WoT + (size_t)o0 * H + kk, H, lane);
                oc = __builtin_amdgcn_wmma_f32_16x16x32_f16(false, af, false, bf, (short)0, oc, false, false);
            }
            const int col = lane & 15, o = o0 + col;
            const float bo = bout[o], mv = mask[g];
            #pragma unroll
            for (int rr = 0; rr < 8; ++rr) {
                const int row = rr + h8;
                node[((size_t)g * N + i0 + row) * O + o] = (oc[rr] + bo) * mv;
            }
        }
    }
}

// ---------------- kernel 5: graph = mean over nodes (node already masked) ----------------
__global__ void k_mean(const float* __restrict__ node, float* __restrict__ graph) {
    __shared__ float sred[256];
    const int g = blockIdx.x, t = threadIdx.x;
    const int o = t & 63, q = t >> 6;
    float s = 0.f;
    for (int i = q; i < N; i += 4) s += node[((size_t)g * N + i) * O + o];
    sred[t] = s;
    __syncthreads();
    if (t < 64) {
        float tot = sred[t] + sred[t + 64] + sred[t + 128] + sred[t + 192];
        graph[(size_t)g * O + t] = tot * (1.0f / (float)N);
    }
}

extern "C" void kernel_launch(void* const* d_in, const int* in_sizes, int n_in,
                              void* d_out, int out_size, void* d_ws, size_t ws_size,
                              hipStream_t stream) {
    const float* x     = (const float*)d_in[0];
    const float* adj   = (const float*)d_in[1];
    const float* mask  = (const float*)d_in[2];
    const float* Win   = (const float*)d_in[3];
    const float* bin   = (const float*)d_in[4];
    const float* Wconv = (const float*)d_in[5];
    const float* bconv = (const float*)d_in[6];
    const float* gamma = (const float*)d_in[7];
    const float* beta  = (const float*)d_in[8];
    const float* rm    = (const float*)d_in[9];
    const float* rv    = (const float*)d_in[10];
    const float* Wout  = (const float*)d_in[11];
    const float* bout  = (const float*)d_in[12];

    char* ws = (char*)d_ws;
    float*     w_dis = (float*)(ws + OFF_DIS);
    float*     w_h0  = (float*)(ws + OFF_H0);
    float*     w_A0  = (float*)(ws + OFF_A0);
    float*     w_B0  = (float*)(ws + OFF_B0);
    float*     w_C1s = (float*)(ws + OFF_C1S);
    float*     w_C1b = (float*)(ws + OFF_C1B);
    _Float16*  w_W1T = (_Float16*)(ws + OFF_W1T);
    _Float16*  w_WoT = (_Float16*)(ws + OFF_WOT);
    float*     w_h1f = (float*)(ws + OFF_H1F32);
    _Float16*  w_h1h = (_Float16*)(ws + OFF_H1F16);
    _Float16*  w_yt  = (_Float16*)(ws + OFF_YT);

    float* node_out  = (float*)d_out;                  // [G,N,O]
    float* graph_out = node_out + (size_t)G * N * O;   // [G,O]

    k_h0    <<<G, H, 0, stream>>>(x, Win, bin, w_h0);
    k_a0    <<<G, H, 0, stream>>>(w_h0, Wconv, gamma, rv, w_A0);
    k_consts<<<(H * H + 255) / 256, 256, 0, stream>>>(Wconv, bconv, gamma, beta, rm, rv, Wout,
                                                      w_B0, w_C1s, w_C1b, w_W1T, w_WoT);
    k_deg   <<<G * N / 8, 256, 0, stream>>>(adj, w_dis);
    k_layer1<<<G * N / 8, 256, 0, stream>>>(adj, w_dis, w_h0, w_A0, w_B0, w_h1f, w_h1h);
    k_ytrans<<<dim3(N / 16, G), 256, 0, stream>>>(w_W1T, w_h1h, w_yt);
    k_layer2<<<dim3(4, G), 256, 0, stream>>>(adj, w_dis, w_yt, w_h1f, w_C1s, w_C1b,
                                             w_WoT, bout, mask, node_out);
    k_mean  <<<G, 256, 0, stream>>>(node_out, graph_out);
}
